// DifferentiableMemoryDictionary_70068096467171
// MI455X (gfx1250) — matrix operand order repaired and checked by reading, not compiled, not run on hardware
//
#include <hip/hip_runtime.h>

// ---------------------------------------------------------------------------
// DifferentiableMemoryDictionary on MI455X (gfx1250, wave32, WMMA)
// B=1024 queries, C=65536 patterns, D=1024, top_k=16.
// v3: double-buffered async global->LDS staging (ASYNCcnt) feeding
// v_wmma_f32_16x16x32_bf16, fused per-tile top-16, candidate reduction +
// softmax-weighted gather.
// ---------------------------------------------------------------------------

typedef __attribute__((ext_vector_type(16))) __bf16 v16bf;
typedef __attribute__((ext_vector_type(8)))  __bf16 v8bf;
typedef __attribute__((ext_vector_type(8)))  float  v8f;
typedef __attribute__((ext_vector_type(4)))  int    v4i;

#define BQ    1024
#define CPAT  65536
#define DIM   1024
#define TOPK  16
#define NTILE 256                 // N columns per GEMM block
#define MTILE 64                  // M rows per GEMM block
#define KSTEP 32
#define NBLK  (CPAT / NTILE)      // 256 column-blocks
#define MBLK  (BQ / MTILE)        // 16 row-blocks
#define EPSN  1e-8f

// ---- gfx1250 async global->LDS path (guarded; falls back to sync copy) ----
#if defined(__AMDGCN__) && __has_builtin(__builtin_amdgcn_global_load_async_to_lds_b128)
#define ASYNC_COPY 1
#else
#define ASYNC_COPY 0
#endif

#if ASYNC_COPY
typedef __attribute__((address_space(1))) v4i as1_v4i;  // global
typedef __attribute__((address_space(3))) v4i as3_v4i;  // LDS
#if __has_builtin(__builtin_amdgcn_s_wait_asynccnt)
#define WAIT_ASYNC(n) __builtin_amdgcn_s_wait_asynccnt(n)
#else
#define WAIT_ASYNC(n) asm volatile("s_wait_asynccnt %0" ::"n"(n))
#endif
#else
#define WAIT_ASYNC(n)
#endif

static __device__ __forceinline__ void copy16_to_lds(const void* g, void* l) {
#if ASYNC_COPY
  __builtin_amdgcn_global_load_async_to_lds_b128((as1_v4i*)g, (as3_v4i*)l, 0, 0);
#else
  *(v8bf*)l = *(const v8bf*)g;
#endif
}

static __device__ __forceinline__ unsigned short f2bf(float f) {
  // round-to-nearest-even float -> bf16 bits
  unsigned int u = __float_as_uint(f);
  u += 0x7FFFu + ((u >> 16) & 1u);
  return (unsigned short)(u >> 16);
}

// ---------------------------------------------------------------------------
// Kernel 1/2: per-row L2 normalize + convert to bf16. One wave per row.
// ---------------------------------------------------------------------------
__global__ __launch_bounds__(256) void rownorm_bf16_kernel(
    const float* __restrict__ src, unsigned short* __restrict__ dst, int rows) {
  const int wave = threadIdx.x >> 5;
  const int lane = threadIdx.x & 31;
  const int row  = blockIdx.x * 8 + wave;
  if (row >= rows) return;

  const float4* r4 = (const float4*)(src + (size_t)row * DIM);
  float4 v[8];
  float ss = 0.0f;
#pragma unroll
  for (int j = 0; j < 8; ++j) {                  // 8 * 32 lanes * 4 = 1024
    v[j] = r4[j * 32 + lane];
    ss += v[j].x * v[j].x + v[j].y * v[j].y + v[j].z * v[j].z + v[j].w * v[j].w;
  }
#pragma unroll
  for (int off = 16; off > 0; off >>= 1) ss += __shfl_xor(ss, off, 32);

  const float scale = 1.0f / fmaxf(sqrtf(ss), EPSN);
  unsigned short* drow = dst + (size_t)row * DIM;
#pragma unroll
  for (int j = 0; j < 8; ++j) {
    ushort4 o;
    o.x = f2bf(v[j].x * scale);
    o.y = f2bf(v[j].y * scale);
    o.z = f2bf(v[j].z * scale);
    o.w = f2bf(v[j].w * scale);
    *(ushort4*)(drow + (size_t)(j * 32 + lane) * 4) = o;
  }
}

// ---------------------------------------------------------------------------
// Kernel 3: WMMA GEMM (64M x 256N per block, K=1024) with double-buffered
// async LDS staging + per-row top-16 within the 256-column tile.
// 8 waves arranged 2(M) x 4(N); each wave: 2x4 tiles of 16x16 WMMA.
//
// LDS (64 KB, unioned):
//   staging: buf0 {A 4KB | B 16KB} @0, buf1 @20KB   (40 KB during K-loop)
//   epilogue: sims f32 [64 x 256]                   (64 KB after K-loop)
// Per wave per stage: 1 (A) + 4 (B) = 5 async b128 instructions -> wait<=5.
// ---------------------------------------------------------------------------
#define LDS_A_OFF   0
#define LDS_B_OFF   4096
#define LDS_BUFSZ   20480

__global__ __launch_bounds__(256, 1) void gemm_topk_kernel(
    const unsigned short* __restrict__ qn, const unsigned short* __restrict__ pb,
    float* __restrict__ cand_val, int* __restrict__ cand_idx) {
  __shared__ __align__(16) char smem[MTILE * NTILE * 4];  // 64 KB

  const int tid   = threadIdx.x;
  const int wave  = tid >> 5;
  const int lane  = tid & 31;
  const int half  = lane >> 4;   // 0 / 1 (lane half, per ISA fragment layout)
  const int l16   = lane & 15;
  const int waveM = wave >> 2;   // 0..1 -> 32 rows each
  const int waveN = wave & 3;    // 0..3 -> 64 cols each

  const int rowBase0 = blockIdx.y * MTILE;               // block's first row
  const int colBase  = blockIdx.x * NTILE;               // block's first col

  const __bf16* Q = (const __bf16*)qn;
  const __bf16* P = (const __bf16*)pb;

  // Per-thread staging assignment:
  //   A tile (64 rows x 64B): thread -> row tid>>2, chunk tid&3 (1 x b128)
  //   B tile (256 rows x 64B): thread -> row tid, chunks 0..3   (4 x b128)
  const int arow = tid >> 2, achk = tid & 3;
  const __bf16* gA = Q + (size_t)(rowBase0 + arow) * DIM + achk * 8;
  const __bf16* gB = P + (size_t)(colBase + tid) * DIM;

  v8f acc[2][4] = {};

  // Prologue: stage 0 into buf0.
  {
    char* la = smem + LDS_A_OFF + arow * 64 + achk * 16;
    char* lb = smem + LDS_B_OFF + tid * 64;
    copy16_to_lds(gA, la);
#pragma unroll
    for (int c = 0; c < 4; ++c) copy16_to_lds(gB + c * 8, lb + c * 16);
  }

  for (int kk = 0; kk < DIM; kk += KSTEP) {
    const int cur = (kk >> 5) & 1;
    // Issue next stage into the other buffer.
    if (kk + KSTEP < DIM) {
      char* la = smem + (cur ^ 1) * LDS_BUFSZ + LDS_A_OFF + arow * 64 + achk * 16;
      char* lb = smem + (cur ^ 1) * LDS_BUFSZ + LDS_B_OFF + tid * 64;
      const int kn = kk + KSTEP;
      copy16_to_lds(gA + kn, la);
      __builtin_prefetch(gB + kn + KSTEP, 0, 0);  // pull next-next B into GL2
#pragma unroll
      for (int c = 0; c < 4; ++c) copy16_to_lds(gB + kn + c * 8, lb + c * 16);
      WAIT_ASYNC(5);   // previous stage (in-order) has landed
    } else {
      WAIT_ASYNC(0);   // final stage fully landed
    }
    __syncthreads();

    const char* Ab = smem + cur * LDS_BUFSZ + LDS_A_OFF;
    const char* Bb = smem + cur * LDS_BUFSZ + LDS_B_OFF;

    v16bf afr[2], bfr[4];
    // A fragment (16x32 MxK): lane<16 holds K {0..7,16..23}, lane>=16 holds
    // K {8..15,24..31} of row M=l16 -> byte chunks half*16 and 32+half*16.
#pragma unroll
    for (int m = 0; m < 2; ++m) {
      const char* arowp = Ab + (waveM * 32 + m * 16 + l16) * 64;
      v8bf a0 = *(const v8bf*)(arowp + half * 16);
      v8bf a1 = *(const v8bf*)(arowp + 32 + half * 16);
      afr[m] = __builtin_shufflevector(a0, a1, 0, 1, 2, 3, 4, 5, 6, 7, 8, 9,
                                       10, 11, 12, 13, 14, 15);
    }
    // B fragment (32x16 KxN): lane holds column N=l16, K {0..7,16..23} for
    // lane<16, K {8..15,24..31} for lane>=16 -> byte chunks half*16 and 32+half*16.
#pragma unroll
    for (int n = 0; n < 4; ++n) {
      const char* browp = Bb + (waveN * 64 + n * 16 + l16) * 64;
      v8bf b0 = *(const v8bf*)(browp + half * 16);
      v8bf b1 = *(const v8bf*)(browp + 32 + half * 16);
      bfr[n] = __builtin_shufflevector(b0, b1, 0, 1, 2, 3, 4, 5, 6, 7, 8, 9,
                                       10, 11, 12, 13, 14, 15);
    }
#pragma unroll
    for (int m = 0; m < 2; ++m)
#pragma unroll
      for (int n = 0; n < 4; ++n)
        acc[m][n] = __builtin_amdgcn_wmma_f32_16x16x32_bf16(
            false, afr[m], false, bfr[n], (short)0, acc[m][n], false, false);
    __syncthreads();  // all waves done reading before buffer reuse
  }

  // Epilogue: C/D layout -> VGPR r, lane: (M = r + 8*half, N = l16)
  float* sims = (float*)smem;  // reuse staging LDS (64 KB)
#pragma unroll
  for (int m = 0; m < 2; ++m)
#pragma unroll
    for (int n = 0; n < 4; ++n)
#pragma unroll
      for (int r = 0; r < 8; ++r) {
        const int rl = waveM * 32 + m * 16 + half * 8 + r;
        const int cl = waveN * 64 + n * 16 + l16;
        sims[rl * NTILE + cl] = acc[m][n][r];
      }
  __syncthreads();

  // Per-row local top-16 over this block's 256 columns (one thread per row).
  if (tid < MTILE) {
    float bv[TOPK];
    int   bi[TOPK];
#pragma unroll
    for (int k = 0; k < TOPK; ++k) { bv[k] = -INFINITY; bi[k] = 0; }
    const float* rowS = sims + tid * NTILE;
    for (int c = 0; c < NTILE; ++c) {
      const float v = rowS[c];
      if (v > bv[TOPK - 1]) {
        bv[TOPK - 1] = v;
        bi[TOPK - 1] = colBase + c;
#pragma unroll
        for (int j = TOPK - 1; j > 0; --j) {
          if (bv[j] > bv[j - 1]) {
            float tv = bv[j]; bv[j] = bv[j - 1]; bv[j - 1] = tv;
            int   ti = bi[j]; bi[j] = bi[j - 1]; bi[j - 1] = ti;
          }
        }
      }
    }
    const int row = rowBase0 + tid;
    const size_t base = ((size_t)row * NBLK + blockIdx.x) * TOPK;
#pragma unroll
    for (int k = 0; k < TOPK; ++k) {
      cand_val[base + k] = bv[k];
      cand_idx[base + k] = bi[k];
    }
  }
}

// ---------------------------------------------------------------------------
// Kernel 4: per query row (1024 blocks): reduce 256*16 candidates to global
// top-16 (sorted desc), complexity-weighted softmax, gather-weighted sum of
// the 16 original f32 pattern rows, write all outputs.
// ---------------------------------------------------------------------------
__global__ __launch_bounds__(256) void topk_finalize_kernel(
    const float* __restrict__ cand_val, const int* __restrict__ cand_idx,
    const float* __restrict__ patterns, const float* __restrict__ comp,
    float* __restrict__ out_mem, float* __restrict__ out_conf,
    float* __restrict__ out_idx, float* __restrict__ out_sims) {
  __shared__ float cv[NBLK * TOPK];  // 4096 floats
  __shared__ int   ci[NBLK * TOPK];
  __shared__ float rv[256];
  __shared__ int   rp[256];
  __shared__ float tv[TOPK];
  __shared__ int   ti[TOPK];
  __shared__ float wk[TOPK];

  const int tid = threadIdx.x;
  const int row = blockIdx.x;
  const size_t base = (size_t)row * NBLK * TOPK;

#pragma unroll
  for (int j = 0; j < 16; ++j) {
    const int p = tid + j * 256;
    cv[p] = cand_val[base + p];
    ci[p] = cand_idx[base + p];
  }
  __syncthreads();

  // 16 rounds of parallel extract-max.
  for (int k = 0; k < TOPK; ++k) {
    float best = -INFINITY;
    int   bpos = 0;
#pragma unroll
    for (int j = 0; j < 16; ++j) {
      const int p = tid + j * 256;
      const float v = cv[p];
      if (v > best) { best = v; bpos = p; }
    }
    rv[tid] = best;
    rp[tid] = bpos;
    __syncthreads();
    for (int off = 128; off > 0; off >>= 1) {
      if (tid < off && rv[tid + off] > rv[tid]) {
        rv[tid] = rv[tid + off];
        rp[tid] = rp[tid + off];
      }
      __syncthreads();
    }
    if (tid == 0) {
      tv[k] = rv[0];
      ti[k] = ci[rp[0]];
      cv[rp[0]] = -INFINITY;
    }
    __syncthreads();
  }

  // weighted_sims = top_sims * (1 + complexity[idx])
  if (tid < TOPK) wk[tid] = tv[tid] * (1.0f + comp[ti[tid]]);
  __syncthreads();

  // Local softmax over 16 (redundant per thread; trivial cost).
  float mx = -INFINITY;
#pragma unroll
  for (int k = 0; k < TOPK; ++k) mx = fmaxf(mx, wk[k]);
  float e[TOPK];
  float sum = 0.0f;
#pragma unroll
  for (int k = 0; k < TOPK; ++k) { e[k] = __expf(wk[k] - mx); sum += e[k]; }
  const float inv = 1.0f / sum;
  float w[TOPK];
  int   til[TOPK];
#pragma unroll
  for (int k = 0; k < TOPK; ++k) { w[k] = e[k] * inv; til[k] = ti[k]; }

  // retrieved_memory[row] = sum_k w[k] * patterns[idx[k]]
  for (int d = tid; d < DIM; d += 256) {
    float a = 0.0f;
#pragma unroll
    for (int k = 0; k < TOPK; ++k)
      a += w[k] * patterns[(size_t)til[k] * DIM + d];
    out_mem[(size_t)row * DIM + d] = a;
  }
  if (tid < TOPK) {
    out_idx[row * TOPK + tid]  = (float)ti[tid];
    out_sims[row * TOPK + tid] = tv[tid];
  }
  if (tid == 0) out_conf[row] = tv[0];  // max of sorted-desc top sims
}

// ---------------------------------------------------------------------------
// Launch. Workspace layout (bytes):
//   qn  bf16 [1024 x 1024]            @ 0          (2 MB)
//   pb  bf16 [65536 x 1024]           @ 2 MB       (128 MB)
//   cand_val f32 [1024 x 256 x 16]    @ 130 MB     (16 MB)
//   cand_idx i32 [1024 x 256 x 16]    @ 146 MB     (16 MB)   total ~162 MB
// Outputs concatenated: mem[1024*1024] | conf[1024] | idx[1024*16] | sims[1024*16]
// ---------------------------------------------------------------------------
extern "C" void kernel_launch(void* const* d_in, const int* in_sizes, int n_in,
                              void* d_out, int out_size, void* d_ws,
                              size_t ws_size, hipStream_t stream) {
  const float* query    = (const float*)d_in[0];
  const float* patterns = (const float*)d_in[1];
  const float* comp     = (const float*)d_in[2];
  // d_in[3] = top_k (fixed 16 per reference setup)

  char* ws = (char*)d_ws;
  unsigned short* qn = (unsigned short*)ws;
  unsigned short* pb = (unsigned short*)(ws + (size_t)BQ * DIM * 2);
  float* cand_val = (float*)(ws + (size_t)BQ * DIM * 2 + (size_t)CPAT * DIM * 2);
  int*   cand_idx = (int*)((char*)cand_val + (size_t)BQ * NBLK * TOPK * 4);

  float* out      = (float*)d_out;
  float* out_mem  = out;
  float* out_conf = out + (size_t)BQ * DIM;
  float* out_idx  = out_conf + BQ;
  float* out_sims = out_idx + (size_t)BQ * TOPK;

  rownorm_bf16_kernel<<<BQ / 8, 256, 0, stream>>>(query, qn, BQ);
  rownorm_bf16_kernel<<<CPAT / 8, 256, 0, stream>>>(patterns, pb, CPAT);

  dim3 grid(NBLK, MBLK);  // 256 x 16
  gemm_topk_kernel<<<grid, 256, 0, stream>>>(qn, pb, cand_val, cand_idx);

  topk_finalize_kernel<<<BQ, 256, 0, stream>>>(cand_val, cand_idx, patterns,
                                               comp, out_mem, out_conf,
                                               out_idx, out_sims);
}
